// PairformerBlock_12661563589134
// MI455X (gfx1250) — matrix operand order repaired and checked by reading, not compiled, not run on hardware
//
#include <hip/hip_runtime.h>
#include <hip/hip_bf16.h>

typedef __bf16 bf16_t;
typedef __attribute__((ext_vector_type(16))) __bf16 v16bf;
typedef __attribute__((ext_vector_type(2)))  __bf16 bf16x2;
typedef __attribute__((ext_vector_type(8)))  float  v8f;
typedef __attribute__((ext_vector_type(4)))  float  f32x4;
typedef __attribute__((ext_vector_type(2)))  float  f32x2;

#define DEV __device__ __forceinline__

static constexpr int Bb  = 2;       // batch
static constexpr int LL  = 320;     // seq len
static constexpr int DD  = 128;     // d_single
static constexpr int DPc = 64;      // d_pair
static constexpr int HH  = 8;       // heads
static constexpr int DHd = 16;      // head dim
static constexpr int DOc = 32;      // outer proj dim
static constexpr int BL  = Bb * LL; // 640 rows

// ---------------------------------------------------------------------------
// WMMA helpers (CDNA5 bf16 16x16x32, fp32 accumulate)
// ---------------------------------------------------------------------------
DEV v8f wmma_bf16(v16bf a, v16bf b, v8f c) {
  return __builtin_amdgcn_wmma_f32_16x16x32_bf16(false, a, false, b, (short)0, c,
                                                 false, false);
}

// A fragment: A row-major [.. x lda], 16x32 tile at (row0, k0).
// Lane l: row m = l&15; lanes 0-15 hold K {0..7,16..23}, lanes 16-31 {8..15,24..31}.
DEV v16bf load_a_frag(const bf16_t* __restrict__ A, int lda, int row0, int k0,
                      int lane) {
  const int m  = lane & 15;
  const int kb = (lane < 16) ? 0 : 8;
  const bf16_t* base = A + (size_t)(row0 + m) * lda + k0;
  v16bf a;
#pragma unroll
  for (int r = 0; r < 4; ++r) {
    bf16x2 p0 = *(const bf16x2*)(base + kb + 2 * r);
    bf16x2 p1 = *(const bf16x2*)(base + 16 + kb + 2 * r);
    a[2 * r]     = p0[0];
    a[2 * r + 1] = p0[1];
    a[8 + 2 * r]     = p1[0];
    a[8 + 2 * r + 1] = p1[1];
  }
  return a;
}

// B fragment from transposed (column-major) weights BT [N x ldbt], 32x16 tile
// at (k0, n0). Lane l: col n = n0+(l&15); lanes 0-15 K {0..7,16..23}, 16-31 {8..15,24..31}.
// Works for both global and LDS-resident BT (compiler picks vmem vs ds ops).
DEV v16bf load_bt_frag(const bf16_t* BT, int ldbt, int k0, int n0, int lane) {
  const int n  = n0 + (lane & 15);
  const int kb = (lane < 16) ? 0 : 8;
  const bf16_t* base = BT + (size_t)n * ldbt + k0 + kb;
  v16bf b;
#pragma unroll
  for (int r = 0; r < 8; ++r) {
    bf16x2 p = *(const bf16x2*)(base + 2 * r + (r < 4 ? 0 : 8));
    b[2 * r]     = p[0];
    b[2 * r + 1] = p[1];
  }
  return b;
}

DEV float wave_reduce_add(float v) {
#pragma unroll
  for (int m = 16; m; m >>= 1) v += __shfl_xor(v, m, 32);
  return v;
}

// ---------------------------------------------------------------------------
// Weight transpose + fp32 -> bf16 convert: dst[n*K+k] = src[k*N+n]
// ---------------------------------------------------------------------------
__global__ void transpose_to_bf16_kernel(const float* __restrict__ src,
                                         bf16_t* __restrict__ dst, int K, int N) {
  int idx = blockIdx.x * blockDim.x + threadIdx.x;
  if (idx >= K * N) return;
  int nn = idx / K;
  int kk = idx - nn * K;
  dst[idx] = (bf16_t)src[(size_t)kk * N + nn];
}

// ---------------------------------------------------------------------------
// LayerNorm over D=128, one wave per row; outputs bf16 activations
// ---------------------------------------------------------------------------
__global__ void layernorm128_kernel(const float* __restrict__ x,
                                    const float* __restrict__ g,
                                    const float* __restrict__ bta,
                                    bf16_t* __restrict__ outB, int rows) {
  const int lane = threadIdx.x & 31;
  const int wid  = threadIdx.x >> 5;
  const int row  = blockIdx.x * (blockDim.x >> 5) + wid;
  if (row >= rows) return;
  f32x4 v = *(const f32x4*)(x + (size_t)row * DD + lane * 4);
  float s  = v[0] + v[1] + v[2] + v[3];
  float sq = v[0] * v[0] + v[1] * v[1] + v[2] * v[2] + v[3] * v[3];
  s  = wave_reduce_add(s);
  sq = wave_reduce_add(sq);
  const float mu   = s * (1.0f / 128.0f);
  const float var  = sq * (1.0f / 128.0f) - mu * mu;
  const float rstd = rsqrtf(var + 1e-5f);
  f32x4 gg = *(const f32x4*)(g + lane * 4);
  f32x4 bb = *(const f32x4*)(bta + lane * 4);
#pragma unroll
  for (int t = 0; t < 4; ++t)
    outB[(size_t)row * DD + lane * 4 + t] =
        (bf16_t)((v[t] - mu) * rstd * gg[t] + bb[t]);
}

// ---------------------------------------------------------------------------
// Generic bf16 WMMA GEMM: out = act(A@B + bias) + residual
// A bf16 row-major [M x K], BT bf16 column-major [N x K].
// Wave computes one 16x16 tile; block = 4 waves over adjacent N-tiles.
// ---------------------------------------------------------------------------
__global__ void gemm_bf16_kernel(const bf16_t* __restrict__ A,
                                 const bf16_t* __restrict__ BT, int M, int N,
                                 int K, const float* __restrict__ bias,
                                 const float* __restrict__ residual,
                                 float* __restrict__ outF,
                                 bf16_t* __restrict__ outB, int act) {
  const int lane = threadIdx.x & 31;
  const int wave = threadIdx.x >> 5;
  const int row0 = blockIdx.x * 16;
  const int n0   = blockIdx.y * 64 + wave * 16;
  if (n0 >= N || row0 >= M) return;

  v8f acc = {};
  for (int k0 = 0; k0 < K; k0 += 32) {
    if (k0 + 64 < K) {  // prefetch two chunks ahead (global_prefetch_b8)
      __builtin_prefetch(A + (size_t)(row0 + (lane & 15)) * K + k0 + 64, 0, 1);
      __builtin_prefetch(BT + (size_t)(n0 + (lane & 15)) * K + k0 + 64, 0, 1);
    }
    v16bf a = load_a_frag(A, K, row0, k0, lane);
    v16bf b = load_bt_frag(BT, K, k0, n0, lane);
    acc = wmma_bf16(a, b, acc);
  }
  const int n  = n0 + (lane & 15);
  const int mb = (lane < 16) ? 0 : 8;
#pragma unroll
  for (int r = 0; r < 8; ++r) {
    const int row = row0 + mb + r;
    float v = acc[r];
    if (bias) v += bias[n];
    if (act) v = 0.5f * v * (1.0f + erff(v * 0.70710678118654752f));
    if (residual) v += residual[(size_t)row * N + n];
    if (outF) outF[(size_t)row * N + n] = v;
    if (outB) outB[(size_t)row * N + n] = (bf16_t)v;
  }
}

// ---------------------------------------------------------------------------
// Pair bias: bias[b,h,i,j] = sum_p z[b,i,j,p] * Wb[p,h].  One block per (b,i);
// 8 waves stride over j; lanes hold 2 p-values, wave-reduce 8 head sums.
// ---------------------------------------------------------------------------
__global__ void pair_bias_kernel(const float* __restrict__ z,
                                 const float* __restrict__ Wb,
                                 float* __restrict__ biasO) {
  __shared__ float wbs[DPc * HH];
  const int t = threadIdx.x;
  for (int idx = t; idx < DPc * HH; idx += blockDim.x) wbs[idx] = Wb[idx];
  __syncthreads();
  const int batch = blockIdx.x / LL;
  const int i     = blockIdx.x - batch * LL;
  const int lane  = t & 31;
  const int wid   = t >> 5;
  for (int j = wid; j < LL; j += 8) {
    const float* zr = z + (((size_t)(batch * LL + i)) * LL + j) * DPc;
    f32x2 zv = *(const f32x2*)(zr + lane * 2);
    float acc[HH];
#pragma unroll
    for (int h = 0; h < HH; ++h)
      acc[h] = zv[0] * wbs[(2 * lane) * HH + h] +
               zv[1] * wbs[(2 * lane + 1) * HH + h];
#pragma unroll
    for (int h = 0; h < HH; ++h) acc[h] = wave_reduce_add(acc[h]);
    if (lane == 0) {
#pragma unroll
      for (int h = 0; h < HH; ++h)
        biasO[(((size_t)batch * HH + h) * LL + i) * LL + j] = acc[h];
    }
  }
}

// ---------------------------------------------------------------------------
// Pair-biased attention: one wave per (b,h,i-tile of 16 rows).
// scores via WMMA (DH=16 zero-padded to K=32), softmax in padded LDS,
// P@V via WMMA over 10 K-chunks. Writes bf16 out at column h*16.
// ---------------------------------------------------------------------------
__global__ void attention_kernel(const float* __restrict__ q,
                                 const float* __restrict__ k,
                                 const float* __restrict__ v,
                                 const float* __restrict__ biasI,
                                 bf16_t* __restrict__ outB) {
  constexpr int SLD = 324;  // padded row stride (324 % 64 == 4 -> no conflicts)
  __shared__ float S[2][16][SLD];
  __shared__ float rsum[2][16];
  const int lane  = threadIdx.x & 31;
  const int w     = threadIdx.x >> 5;
  const int it    = blockIdx.x * 2 + w;  // 0..19
  const int h     = blockIdx.y;
  const int batch = blockIdx.z;
  const int i0    = it * 16;
  const int row0  = batch * LL + i0;
  const int m     = lane & 15;
  const int kb    = (lane < 16) ? 0 : 8;

  // Q fragment, K padded 16 -> 32 with zeros
  v16bf aq;
  {
    const float* qr = q + (size_t)(row0 + m) * DD + h * DHd;
#pragma unroll
    for (int r = 0; r < 4; ++r) {
      aq[2 * r]     = (bf16_t)qr[kb + 2 * r];
      aq[2 * r + 1] = (bf16_t)qr[kb + 2 * r + 1];
      aq[8 + 2 * r]     = (bf16_t)0.0f;
      aq[8 + 2 * r + 1] = (bf16_t)0.0f;
    }
  }
  const float* biasRow = biasI + (((size_t)batch * HH + h) * LL) * LL;

  // scores + bias -> LDS
  for (int jt = 0; jt < LL / 16; ++jt) {
    const int j0 = jt * 16;
    v16bf bk;
#pragma unroll
    for (int e = 0; e < 16; ++e) bk[e] = (bf16_t)0.0f;
    {  // K-rows 16..31 of B are the zero padding
      const float* kr = k + (size_t)(batch * LL + j0 + m) * DD + h * DHd + kb;
#pragma unroll
      for (int r = 0; r < 4; ++r) {
        f32x2 kv = *(const f32x2*)(kr + 2 * r);
        bk[2 * r]     = (bf16_t)kv[0];
        bk[2 * r + 1] = (bf16_t)kv[1];
      }
    }
    v8f c = {};
    c = wmma_bf16(aq, bk, c);
    const int n  = lane & 15;
    const int mb = (lane < 16) ? 0 : 8;
#pragma unroll
    for (int r = 0; r < 8; ++r) {
      const int mm = mb + r;
      S[w][mm][j0 + n] =
          c[r] * 0.25f + biasRow[(size_t)(i0 + mm) * LL + (j0 + n)];
    }
  }

  // row softmax: 2 lanes per row (halves), combine with shfl_xor(16)
  const int half = lane >> 4;
  const int mr   = lane & 15;
  float mx = -3.4e38f;
  for (int jj = half * 160; jj < half * 160 + 160; ++jj)
    mx = fmaxf(mx, S[w][mr][jj]);
  mx = fmaxf(mx, __shfl_xor(mx, 16, 32));
  float sm = 0.0f;
  for (int jj = half * 160; jj < half * 160 + 160; ++jj) {
    const float e = __expf(S[w][mr][jj] - mx);
    S[w][mr][jj]  = e;
    sm += e;
  }
  sm += __shfl_xor(sm, 16, 32);
  rsum[w][mr] = sm;  // both halves write identical value

  // P @ V
  v8f acc = {};
  for (int c = 0; c < LL / 32; ++c) {
    const int j0 = c * 32;
    v16bf ap;
#pragma unroll
    for (int r = 0; r < 4; ++r) {
      ap[2 * r]     = (bf16_t)S[w][m][j0 + kb + 2 * r];
      ap[2 * r + 1] = (bf16_t)S[w][m][j0 + kb + 2 * r + 1];
      ap[8 + 2 * r]     = (bf16_t)S[w][m][j0 + 16 + kb + 2 * r];
      ap[8 + 2 * r + 1] = (bf16_t)S[w][m][j0 + 16 + kb + 2 * r + 1];
    }
    v16bf bv;
    {
      const int n   = lane & 15;
      const int kb2 = (lane < 16) ? 0 : 8;
#pragma unroll
      for (int r = 0; r < 8; ++r) {
        const int jg = batch * LL + j0 + kb2 + 2 * r + (r < 4 ? 0 : 8);
        bv[2 * r]     = (bf16_t)v[(size_t)jg * DD + h * DHd + n];
        bv[2 * r + 1] = (bf16_t)v[(size_t)(jg + 1) * DD + h * DHd + n];
      }
    }
    acc = wmma_bf16(ap, bv, acc);
  }
  const int n  = lane & 15;
  const int mb = (lane < 16) ? 0 : 8;
#pragma unroll
  for (int r = 0; r < 8; ++r) {
    const int mm  = mb + r;
    const float o = acc[r] / rsum[w][mm];
    outB[(size_t)(row0 + mm) * DD + h * DHd + n] = (bf16_t)o;
  }
}

// ---------------------------------------------------------------------------
// Outer-product pair update + residual + LayerNorm, fused into final z.
// Block = 8 waves, wave per (b, i, j-tile of 16). The whole WzT (bf16, padded
// row stride) is staged once per block into LDS (CDNA5: 320KB LDS per WGP) so
// all 8 waves hit LDS instead of streaming 128KB each from L2.
// K-chunk kk covers exactly c=kk (32 | 1024) -> A-frag = p1[i,kk] * fixed p2
// vector, built in VALU (co-executes with the XDL wmma pipe).
// ---------------------------------------------------------------------------
static constexpr int WZ_LDS_STRIDE = 1032;  // bf16 elems; 516 dwords % 64 = 4
static constexpr size_t WZ_LDS_BYTES =
    (size_t)DPc * WZ_LDS_STRIDE * 2;                       // 132096
static constexpr size_t U_STRIDE = 66;                     // padded p-stride
static constexpr size_t U_BYTES  = 8 * 16 * U_STRIDE * 4;  // 33792
static constexpr size_t PAIR_LDS_BYTES = WZ_LDS_BYTES + U_BYTES;

__global__ void pair_update_kernel(const float* __restrict__ zin,
                                   const float* __restrict__ p1,
                                   const float* __restrict__ p2,
                                   const bf16_t* __restrict__ WzT,  // [64][1024]
                                   const float* __restrict__ bz,
                                   const float* __restrict__ lg,
                                   const float* __restrict__ lb,
                                   float* __restrict__ zout) {
  extern __shared__ char smem[];
  bf16_t* wzL = (bf16_t*)smem;
  float*  U   = (float*)(smem + WZ_LDS_BYTES);  // [8 waves][16 j][66]

  // ---- stage WzT (64 x 1024 bf16) into LDS with padded stride ----
  {
    const uint32_t* src = (const uint32_t*)WzT;  // 64*512 dwords
    uint32_t* dst = (uint32_t*)smem;
    for (int idx = threadIdx.x; idx < DPc * 512; idx += 256) {
      const int n  = idx >> 9;
      const int kd = idx & 511;
      if (idx + 2048 < DPc * 512) __builtin_prefetch(src + idx + 2048, 0, 1);
      dst[n * (WZ_LDS_STRIDE / 2) + kd] = src[idx];
    }
  }
  __syncthreads();

  const int lane  = threadIdx.x & 31;
  const int w     = threadIdx.x >> 5;
  const int jt    = blockIdx.x * 8 + w;  // 0..23; >=20 idle
  if (jt >= LL / 16) return;
  const int i     = blockIdx.y;
  const int batch = blockIdx.z;
  const int j0    = jt * 16;
  const int m     = lane & 15;
  const int kb    = (lane < 16) ? 0 : 8;
  float* Uw = U + (size_t)w * 16 * U_STRIDE;

  // fixed per-lane p2 values arranged in A-fragment order
  const float* p2r = p2 + (size_t)(batch * LL + j0 + m) * DOc;
  float p2v[16];
#pragma unroll
  for (int r = 0; r < 8; ++r) {
    p2v[r]     = p2r[kb + r];
    p2v[8 + r] = p2r[16 + kb + r];
  }
  const float* p1r = p1 + (size_t)(batch * LL + i) * DOc;

  v8f acc0 = {}, acc1 = {}, acc2 = {}, acc3 = {};
  for (int kk = 0; kk < 32; ++kk) {
    const float s = p1r[kk];  // wave-uniform scalar
    v16bf a;
#pragma unroll
    for (int e = 0; e < 16; ++e) a[e] = (bf16_t)(s * p2v[e]);
    const int k0 = kk * 32;
    v16bf b0 = load_bt_frag(wzL, WZ_LDS_STRIDE, k0, 0, lane);
    v16bf b1 = load_bt_frag(wzL, WZ_LDS_STRIDE, k0, 16, lane);
    v16bf b2 = load_bt_frag(wzL, WZ_LDS_STRIDE, k0, 32, lane);
    v16bf b3 = load_bt_frag(wzL, WZ_LDS_STRIDE, k0, 48, lane);
    acc0 = wmma_bf16(a, b0, acc0);
    acc1 = wmma_bf16(a, b1, acc1);
    acc2 = wmma_bf16(a, b2, acc2);
    acc3 = wmma_bf16(a, b3, acc3);
  }

  // epilogue: +bz into LDS tile [16 j][64 p]
  const int nn = lane & 15;
  const int mb = (lane < 16) ? 0 : 8;
#pragma unroll
  for (int r = 0; r < 8; ++r) {
    float* Ur = Uw + (size_t)(mb + r) * U_STRIDE;
    Ur[nn]      = acc0[r] + bz[nn];
    Ur[16 + nn] = acc1[r] + bz[16 + nn];
    Ur[32 + nn] = acc2[r] + bz[32 + nn];
    Ur[48 + nn] = acc3[r] + bz[48 + nn];
  }

  // residual + LayerNorm over p (64), 2 values per lane
  const int p0 = lane * 2;
  const float g0 = lg[p0], g1 = lg[p0 + 1];
  const float b0f = lb[p0], b1f = lb[p0 + 1];
  for (int jj = 0; jj < 16; ++jj) {
    const size_t zoff =
        (((size_t)(batch * LL + i)) * LL + (j0 + jj)) * DPc;
    const float* Ur = Uw + (size_t)jj * U_STRIDE;
    const float v0 = zin[zoff + p0] + Ur[p0];
    const float v1 = zin[zoff + p0 + 1] + Ur[p0 + 1];
    float s  = v0 + v1;
    float sq = v0 * v0 + v1 * v1;
    s  = wave_reduce_add(s);
    sq = wave_reduce_add(sq);
    const float mu   = s * (1.0f / 64.0f);
    const float var  = sq * (1.0f / 64.0f) - mu * mu;
    const float rstd = rsqrtf(var + 1e-5f);
    f32x2 o;
    o[0] = (v0 - mu) * rstd * g0 + b0f;
    o[1] = (v1 - mu) * rstd * g1 + b1f;
    *(f32x2*)(zout + zoff + p0) = o;
  }
}

// ---------------------------------------------------------------------------
// Host-side pipeline
// ---------------------------------------------------------------------------
extern "C" void kernel_launch(void* const* d_in, const int* in_sizes, int n_in,
                              void* d_out, int out_size, void* d_ws,
                              size_t ws_size, hipStream_t stream) {
  const float* x    = (const float*)d_in[0];
  const float* z    = (const float*)d_in[1];
  const float* Wq   = (const float*)d_in[2];
  const float* Wk   = (const float*)d_in[3];
  const float* Wv   = (const float*)d_in[4];
  const float* Wb   = (const float*)d_in[5];
  const float* Wo   = (const float*)d_in[6];
  const float* bo   = (const float*)d_in[7];
  const float* ln1g = (const float*)d_in[8];
  const float* ln1b = (const float*)d_in[9];
  const float* W1   = (const float*)d_in[10];
  const float* b1   = (const float*)d_in[11];
  const float* W2   = (const float*)d_in[12];
  const float* b2   = (const float*)d_in[13];
  const float* ln2g = (const float*)d_in[14];
  const float* ln2b = (const float*)d_in[15];
  const float* Wp1  = (const float*)d_in[16];
  const float* bp1  = (const float*)d_in[17];
  const float* Wp2  = (const float*)d_in[18];
  const float* bp2  = (const float*)d_in[19];
  const float* Wz   = (const float*)d_in[20];
  const float* bz   = (const float*)d_in[21];
  const float* lnpg = (const float*)d_in[22];
  const float* lnpb = (const float*)d_in[23];

  float* outX = (float*)d_out;             // [640 x 128]
  float* outZ = (float*)d_out + BL * DD;   // [2 x 320 x 320 x 64]

  char* ws   = (char*)d_ws;
  size_t off = 0;
  auto take = [&](size_t bytes) -> char* {
    off = (off + 255) & ~(size_t)255;
    char* p = ws + off;
    off += bytes;
    return p;
  };

  bf16_t* WqT  = (bf16_t*)take(128 * 128 * 2);
  bf16_t* WkT  = (bf16_t*)take(128 * 128 * 2);
  bf16_t* WvT  = (bf16_t*)take(128 * 128 * 2);
  bf16_t* WoT  = (bf16_t*)take(128 * 128 * 2);
  bf16_t* W1T  = (bf16_t*)take(128 * 512 * 2);
  bf16_t* W2T  = (bf16_t*)take(512 * 128 * 2);
  bf16_t* Wp1T = (bf16_t*)take(128 * 32 * 2);
  bf16_t* Wp2T = (bf16_t*)take(128 * 32 * 2);
  bf16_t* WzT  = (bf16_t*)take(1024 * 64 * 2);
  bf16_t* xnB  = (bf16_t*)take((size_t)BL * DD * 2);
  float*  qf   = (float*)take((size_t)BL * DD * 4);
  float*  kf   = (float*)take((size_t)BL * DD * 4);
  float*  vf   = (float*)take((size_t)BL * DD * 4);
  float*  biasB = (float*)take((size_t)Bb * HH * LL * LL * 4);
  bf16_t* aoB  = (bf16_t*)take((size_t)BL * DD * 2);
  float*  x1f  = (float*)take((size_t)BL * DD * 4);
  bf16_t* xn2B = (bf16_t*)take((size_t)BL * DD * 2);
  bf16_t* hB   = (bf16_t*)take((size_t)BL * 512 * 2);
  bf16_t* x2B  = (bf16_t*)take((size_t)BL * DD * 2);
  float*  p1f  = (float*)take((size_t)BL * DOc * 4);
  float*  p2f  = (float*)take((size_t)BL * DOc * 4);
  (void)ws_size; (void)in_sizes; (void)n_in; (void)out_size;

  auto tr = [&](const float* src, bf16_t* dst, int K, int N) {
    int total = K * N;
    transpose_to_bf16_kernel<<<(total + 255) / 256, 256, 0, stream>>>(src, dst,
                                                                      K, N);
  };
  tr(Wq, WqT, 128, 128);
  tr(Wk, WkT, 128, 128);
  tr(Wv, WvT, 128, 128);
  tr(Wo, WoT, 128, 128);
  tr(W1, W1T, 128, 512);
  tr(W2, W2T, 512, 128);
  tr(Wp1, Wp1T, 128, 32);
  tr(Wp2, Wp2T, 128, 32);
  tr(Wz, WzT, 1024, 64);

  // LN1
  layernorm128_kernel<<<BL / 4, 128, 0, stream>>>(x, ln1g, ln1b, xnB, BL);

  auto gemm = [&](const bf16_t* A, const bf16_t* BT, int M, int N, int K,
                  const float* bias, const float* res, float* oF, bf16_t* oB,
                  int act) {
    dim3 grid(M / 16, (N + 63) / 64);
    gemm_bf16_kernel<<<grid, 128, 0, stream>>>(A, BT, M, N, K, bias, res, oF,
                                               oB, act);
  };

  // QKV projections
  gemm(xnB, WqT, BL, DD, DD, nullptr, nullptr, qf, nullptr, 0);
  gemm(xnB, WkT, BL, DD, DD, nullptr, nullptr, kf, nullptr, 0);
  gemm(xnB, WvT, BL, DD, DD, nullptr, nullptr, vf, nullptr, 0);

  // pair bias projection
  pair_bias_kernel<<<Bb * LL, 256, 0, stream>>>(z, Wb, biasB);

  // attention
  attention_kernel<<<dim3(LL / 32, HH, Bb), 64, 0, stream>>>(qf, kf, vf, biasB,
                                                             aoB);

  // output projection + residual: x1 = x + out@Wo + bo
  gemm(aoB, WoT, BL, DD, DD, bo, x, x1f, nullptr, 0);

  // FFN
  layernorm128_kernel<<<BL / 4, 128, 0, stream>>>(x1f, ln2g, ln2b, xn2B, BL);
  gemm(xn2B, W1T, BL, 512, DD, b1, nullptr, nullptr, hB, 1);  // gelu
  gemm(hB, W2T, BL, DD, 512, b2, x1f, outX, x2B, 0);          // final x

  // outer projections
  gemm(x2B, Wp1T, BL, DOc, DD, bp1, nullptr, p1f, nullptr, 0);
  gemm(x2B, Wp2T, BL, DOc, DD, bp2, nullptr, p2f, nullptr, 0);

  // pair update + LN -> final z (8 waves/block, WzT staged in LDS)
  pair_update_kernel<<<dim3((LL / 16 + 7) / 8, LL, Bb), 256, PAIR_LDS_BYTES,
                       stream>>>(z, p1f, p2f, WzT, bz, lnpg, lnpb, outZ);
}
